// UniLMEncoder_29678224015885
// MI455X (gfx1250) — compile-verified, matrix operand order checked by reading, not run on hardware
//
#include <hip/hip_runtime.h>
#include <hip/hip_bf16.h>
#include <math.h>

// Problem constants: N,L,M,D,H,FF,V,P,LAYERS = 4,256,2,768,12,3072,32000,1024,4
#define NN    4
#define LL    256
#define MM    2
#define DD    768
#define HH    12
#define FFD   3072
#define DHH   64          // D / H
#define NMB   8           // N*M (memory rows)
#define TT    1032        // NMB + N*L total token rows
#define KKEY  258         // M + L keys per token
#define NLAYERS 4

typedef __attribute__((ext_vector_type(16))) __bf16    v16bf;
typedef __attribute__((ext_vector_type(8)))  __bf16    v8bf;
typedef __attribute__((ext_vector_type(8)))  float     v8f;
typedef __attribute__((ext_vector_type(4)))  float     v4f;
typedef __attribute__((ext_vector_type(4)))  unsigned  v4u;

// two f32 -> packed bf16 pair (round-half-up), low word = a.
// 2x v_add + 1x v_perm_b32.
__device__ __forceinline__ unsigned pack2bf(float a, float b) {
  unsigned ua = __builtin_bit_cast(unsigned, a) + 0x8000u;
  unsigned ub = __builtin_bit_cast(unsigned, b) + 0x8000u;
  return __builtin_amdgcn_perm(ub, ua, 0x07060302u);   // (ua>>16)|(ub&0xFFFF0000)
}

__device__ __forceinline__ float gelu_exact(float x) {
  return 0.5f * x * (1.0f + erff(x * 0.70710678118654752f));
}

// ---------------------------------------------------------------------------
// Embedding: x[t][d] = memory+pos (t < NMB) or emb_table[id]+pos (tokens)
// ---------------------------------------------------------------------------
__global__ __launch_bounds__(256)
void embed_kernel(const int* __restrict__ ids, const float* __restrict__ memory,
                  const float* __restrict__ emb, const float* __restrict__ pos,
                  float* __restrict__ x)
{
  const int t = blockIdx.x;
  const int tid = threadIdx.x;
  #pragma unroll
  for (int j = 0; j < 3; ++j) {
    int d = tid + j * 256;
    float val;
    if (t < NMB) {                       // memory slot: row t = n*M+m, pos index m
      int m = t % MM;
      val = memory[(size_t)t * DD + d] + pos[(size_t)m * DD + d];
    } else {
      int jj = t - NMB;
      int n = jj / LL, l = jj % LL;
      int id = ids[n * LL + l];
      val = emb[(size_t)id * DD + d] + pos[(size_t)(MM + l) * DD + d];
    }
    x[(size_t)t * DD + d] = val;
  }
}

// ---------------------------------------------------------------------------
// Core tiled WMMA GEMM: C[M x Nd] = A[M x K] * B[K x Nd] + bias (+epilogue)
// 256 threads = 8 wave32. Block tile 128x64; wave tile 32x32 (4 WMMA accums).
// Software-pipelined: register prefetch of tile k+1 during compute of tile k,
// LDS ping-pong buffers -> single barrier per K-step.
// epi: 0 = bias, 1 = bias+exact GELU, 2 = bias+residual
// ---------------------------------------------------------------------------
__device__ __forceinline__
void gemm_core(__bf16 (*As)[128][40], __bf16 (*Bt)[64][40],
               const float* __restrict__ A, const float* __restrict__ B,
               const float* __restrict__ bias, const float* __restrict__ resid,
               float* __restrict__ C, int Mrows, int K, int Nd, int epi)
{
  const int tid  = threadIdx.x;
  const int lane = tid & 31;
  const int w    = tid >> 5;      // wave 0..7
  const int wm   = w & 3;         // wave row group (32 rows each)
  const int wn   = w >> 2;        // wave col group (32 cols each)
  const int half = lane >> 4;     // WMMA 16-bit operand lane-half
  const int l16  = lane & 15;

  const int mblk = blockIdx.x * 128;
  const int nblk = blockIdx.y * 64;

  v8f acc[2][2] = {{{}, {}}, {{}, {}}};   // [mi][ni]

  // A staging: row ar = tid>>1 (0..127), k-group ac = (tid&1)*16
  const int ar  = tid >> 1;
  const int ac  = (tid & 1) << 4;
  const int agr = mblk + ar;
  const bool aok = (agr < Mrows);
  const float* aRow = A + (size_t)(aok ? agr : 0) * K;

  // B staging: column n = tid&63, k-group bk0 = (tid>>6)*8
  const int bn  = tid & 63;
  const int bk0 = (tid >> 6) << 3;
  const float* bCol = B + nblk + bn;

  // zero-fill LDS rows for out-of-range A rows once (both buffers)
  if (!aok) {
    v4u z = {0u, 0u, 0u, 0u};
    *(v4u*)&As[0][ar][ac]     = z;  *(v4u*)&As[0][ar][ac + 8] = z;
    *(v4u*)&As[1][ar][ac]     = z;  *(v4u*)&As[1][ar][ac + 8] = z;
  }

  // staged registers (tile currently being fetched from global)
  v4f fa0, fa1, fa2, fa3;
  float fb[8];

  auto load_tile = [&](int kk) {
    if (aok) {
      const v4f* ap = (const v4f*)(aRow + kk + ac);
      fa0 = ap[0]; fa1 = ap[1]; fa2 = ap[2]; fa3 = ap[3];
    }
    #pragma unroll
    for (int j = 0; j < 8; ++j)
      fb[j] = bCol[(size_t)(kk + bk0 + j) * Nd];
  };

  load_tile(0);

  for (int kk = 0; kk < K; kk += 32) {
    const int pb = (kk >> 5) & 1;
    // ---- pack + store staged tile into LDS buffer pb ----
    if (aok) {
      v4u p0, p1;
      p0[0] = pack2bf(fa0[0], fa0[1]); p0[1] = pack2bf(fa0[2], fa0[3]);
      p0[2] = pack2bf(fa1[0], fa1[1]); p0[3] = pack2bf(fa1[2], fa1[3]);
      p1[0] = pack2bf(fa2[0], fa2[1]); p1[1] = pack2bf(fa2[2], fa2[3]);
      p1[2] = pack2bf(fa3[0], fa3[1]); p1[3] = pack2bf(fa3[2], fa3[3]);
      *(v4u*)&As[pb][ar][ac]     = p0;
      *(v4u*)&As[pb][ar][ac + 8] = p1;
    }
    {
      v4u p;
      p[0] = pack2bf(fb[0], fb[1]); p[1] = pack2bf(fb[2], fb[3]);
      p[2] = pack2bf(fb[4], fb[5]); p[3] = pack2bf(fb[6], fb[7]);
      *(v4u*)&Bt[pb][bn][bk0] = p;
    }
    __syncthreads();

    // ---- prefetch next tile into registers (overlaps with WMMA below) ----
    if (kk + 32 < K) load_tile(kk + 32);

    // ---- fragments: lane(l16,half) K-layout [8h,8h+8) U [16+8h,16+8h+8) ----
    v16bf bfr[2];
    #pragma unroll
    for (int ni = 0; ni < 2; ++ni) {
      const int brow = wn * 32 + ni * 16 + l16;
      v8bf blo = *(const v8bf*)&Bt[pb][brow][8 * half];
      v8bf bhi = *(const v8bf*)&Bt[pb][brow][16 + 8 * half];
      #pragma unroll
      for (int i = 0; i < 8; ++i) { bfr[ni][i] = blo[i]; bfr[ni][8 + i] = bhi[i]; }
    }
    #pragma unroll
    for (int mi = 0; mi < 2; ++mi) {
      const int arow = wm * 32 + mi * 16 + l16;
      v8bf alo = *(const v8bf*)&As[pb][arow][8 * half];
      v8bf ahi = *(const v8bf*)&As[pb][arow][16 + 8 * half];
      v16bf a;
      #pragma unroll
      for (int i = 0; i < 8; ++i) { a[i] = alo[i]; a[8 + i] = ahi[i]; }
      acc[mi][0] = __builtin_amdgcn_wmma_f32_16x16x32_bf16(false, a, false, bfr[0], (short)0, acc[mi][0], false, false);
      acc[mi][1] = __builtin_amdgcn_wmma_f32_16x16x32_bf16(false, a, false, bfr[1], (short)0, acc[mi][1], false, false);
    }
    // No trailing barrier: ping-pong buffers + the barrier above guarantee
    // all waves finished reading buffer pb before it is overwritten (iter+2).
  }

  // ---- epilogue: C/D layout — VGPR v, lane(l16,half) => (m=v+8*half, n=l16) ----
  auto store_all = [&](bool guard) {
    #pragma unroll
    for (int ni = 0; ni < 2; ++ni) {
      const int col = nblk + wn * 32 + ni * 16 + l16;
      const float bs = bias[col];
      #pragma unroll
      for (int mi = 0; mi < 2; ++mi) {
        #pragma unroll
        for (int vi = 0; vi < 8; ++vi) {
          const int row = mblk + wm * 32 + mi * 16 + 8 * half + vi;
          if (guard && row >= Mrows) continue;
          float x0 = acc[mi][ni][vi] + bs;
          if (epi == 1)      x0 = gelu_exact(x0);
          else if (epi == 2) x0 += resid[(size_t)row * Nd + col];
          C[(size_t)row * Nd + col] = x0;
        }
      }
    }
  };
  if (mblk + 128 <= Mrows) store_all(false);
  else                     store_all(true);
}

__global__ __launch_bounds__(256)
void gemm_bias_epi(const float* __restrict__ A, const float* __restrict__ B,
                   const float* __restrict__ bias, const float* __restrict__ resid,
                   float* __restrict__ C, int Mrows, int K, int Nd, int epi)
{
  __shared__ alignas(16) __bf16 As[2][128][40];
  __shared__ alignas(16) __bf16 Bt[2][64][40];
  gemm_core(As, Bt, A, B, bias, resid, C, Mrows, K, Nd, epi);
}

// Fused Q/K/V projection: same A, three weight/bias/output sets via blockIdx.z
struct QKVArgs {
  const float* B0; const float* B1; const float* B2;
  const float* bias0; const float* bias1; const float* bias2;
  float* C0; float* C1; float* C2;
};

__global__ __launch_bounds__(256)
void gemm_qkv(const float* __restrict__ A, QKVArgs args, int Mrows, int K, int Nd)
{
  __shared__ alignas(16) __bf16 As[2][128][40];
  __shared__ alignas(16) __bf16 Bt[2][64][40];
  const int z = blockIdx.z;
  const float* B    = (z == 0) ? args.B0    : (z == 1) ? args.B1    : args.B2;
  const float* bias = (z == 0) ? args.bias0 : (z == 1) ? args.bias1 : args.bias2;
  float*       C    = (z == 0) ? args.C0    : (z == 1) ? args.C1    : args.C2;
  gemm_core(As, Bt, A, B, bias, nullptr, C, Mrows, K, Nd, 0);
}

// ---------------------------------------------------------------------------
// Attention: one block per (token, head). 258-key softmax attention within
// the token's sentence (M memory slots + L sentence tokens).
// ---------------------------------------------------------------------------
__global__ __launch_bounds__(256)
void attn_kernel(const float* __restrict__ q, const float* __restrict__ k,
                 const float* __restrict__ v, float* __restrict__ ctx)
{
  const int t = blockIdx.x;
  const int h = blockIdx.y;
  const int tid = threadIdx.x;

  __shared__ float qs[DHH];
  __shared__ float sc[KKEY];
  __shared__ float rbuf[256];

  const int n = (t < NMB) ? (t / MM) : ((t - NMB) / LL);
  const int tokBase = NMB + n * LL;     // first token row of this sentence

  if (tid < DHH) qs[tid] = q[(size_t)t * DD + h * DHH + tid];
  __syncthreads();

  // ---- scores + local max: key 'tid'; threads 0,1 also do keys 256,257 ----
  float lmax;
  {
    const int kr = (tid < MM) ? (n * MM + tid) : (tokBase + tid - MM);
    const float* kp = k + (size_t)kr * DD + h * DHH;
    float acc = 0.f;
    #pragma unroll 8
    for (int d = 0; d < DHH; ++d) acc += qs[d] * kp[d];
    acc *= 0.125f;                      // 1/sqrt(64)
    sc[tid] = acc;
    lmax = acc;
  }
  if (tid < KKEY - 256) {
    const int ki = 256 + tid;
    const float* kp = k + (size_t)(tokBase + ki - MM) * DD + h * DHH;
    float acc = 0.f;
    #pragma unroll 8
    for (int d = 0; d < DHH; ++d) acc += qs[d] * kp[d];
    acc *= 0.125f;
    sc[ki] = acc;
    lmax = fmaxf(lmax, acc);
  }
  rbuf[tid] = lmax; __syncthreads();
  #pragma unroll
  for (int s = 128; s > 0; s >>= 1) {
    if (tid < s) rbuf[tid] = fmaxf(rbuf[tid], rbuf[tid + s]);
    __syncthreads();
  }
  float mx = rbuf[0];
  __syncthreads();

  // ---- exp + sum ----
  float lsum;
  {
    float e = expf(sc[tid] - mx);
    sc[tid] = e; lsum = e;
  }
  if (tid < KKEY - 256) {
    float e = expf(sc[256 + tid] - mx);
    sc[256 + tid] = e; lsum += e;
  }
  rbuf[tid] = lsum; __syncthreads();
  #pragma unroll
  for (int s = 128; s > 0; s >>= 1) {
    if (tid < s) rbuf[tid] += rbuf[tid + s];
    __syncthreads();
  }
  float inv = 1.0f / rbuf[0];
  __syncthreads();

  // ---- ctx = probs @ V ----
  if (tid < DHH) {
    const int off = h * DHH + tid;
    float acc = sc[0] * v[(size_t)(n * MM + 0) * DD + off]
              + sc[1] * v[(size_t)(n * MM + 1) * DD + off];
    const float* vp = v + (size_t)tokBase * DD + off;
    #pragma unroll 4
    for (int l = 0; l < LL; ++l)
      acc += sc[MM + l] * vp[(size_t)l * DD];
    ctx[(size_t)t * DD + off] = acc * inv;
  }
}

// ---------------------------------------------------------------------------
// out = LayerNorm(a + badd?) * g + be      (one block per row, D=768)
// ---------------------------------------------------------------------------
__global__ __launch_bounds__(256)
void add_ln_kernel(const float* __restrict__ a, const float* __restrict__ badd,
                   const float* __restrict__ g, const float* __restrict__ be,
                   float* __restrict__ out)
{
  const int row = blockIdx.x;
  const int tid = threadIdx.x;
  __shared__ float rbuf[256];
  float loc[3];
  float s = 0.f;
  #pragma unroll
  for (int j = 0; j < 3; ++j) {
    int d = tid + j * 256;
    float x = a[(size_t)row * DD + d];
    if (badd) x += badd[(size_t)row * DD + d];
    loc[j] = x; s += x;
  }
  rbuf[tid] = s; __syncthreads();
  #pragma unroll
  for (int st = 128; st > 0; st >>= 1) {
    if (tid < st) rbuf[tid] += rbuf[tid + st];
    __syncthreads();
  }
  float mu = rbuf[0] * (1.0f / DD);
  __syncthreads();
  float ss = 0.f;
  #pragma unroll
  for (int j = 0; j < 3; ++j) { float dx = loc[j] - mu; ss += dx * dx; }
  rbuf[tid] = ss; __syncthreads();
  #pragma unroll
  for (int st = 128; st > 0; st >>= 1) {
    if (tid < st) rbuf[tid] += rbuf[tid + st];
    __syncthreads();
  }
  float rstd = rsqrtf(rbuf[0] * (1.0f / DD) + 1e-5f);
  #pragma unroll
  for (int j = 0; j < 3; ++j) {
    int d = tid + j * 256;
    out[(size_t)row * DD + d] = (loc[j] - mu) * rstd * g[d] + be[d];
  }
}

// ---------------------------------------------------------------------------
extern "C" void kernel_launch(void* const* d_in, const int* in_sizes, int n_in,
                              void* d_out, int out_size, void* d_ws, size_t ws_size,
                              hipStream_t stream)
{
  const int*   ids = (const int*)  d_in[0];
  const float* mem = (const float*)d_in[1];
  const float* emb = (const float*)d_in[2];
  const float* pos = (const float*)d_in[3];
  const float* Wq  = (const float*)d_in[4];
  const float* bq  = (const float*)d_in[5];
  const float* Wk  = (const float*)d_in[6];
  const float* bk  = (const float*)d_in[7];
  const float* Wv  = (const float*)d_in[8];
  const float* bv  = (const float*)d_in[9];
  const float* W1  = (const float*)d_in[10];
  const float* b1  = (const float*)d_in[11];
  const float* W2  = (const float*)d_in[12];
  const float* b2  = (const float*)d_in[13];
  const float* g1  = (const float*)d_in[14];
  const float* be1 = (const float*)d_in[15];
  const float* g2  = (const float*)d_in[16];
  const float* be2 = (const float*)d_in[17];

  const size_t TD = (size_t)TT * DD;
  float* xb = (float*)d_ws;      // x           [T,D]
  float* qb = xb + TD;           // q           [T,D]
  float* kb = qb + TD;           // k           [T,D]
  float* vb = kb + TD;           // v           [T,D]
  float* cb = vb + TD;           // attn ctx    [T,D]
  float* tb = cb + TD;           // ffn2 out    [T,D]
  float* hb = tb + TD;           // ffn hidden  [T,FF]

  dim3 blk(256);
  embed_kernel<<<dim3(TT), blk, 0, stream>>>(ids, mem, emb, pos, xb);

  dim3 gQKV((TT + 127) / 128, DD / 64, 3);   // 9 x 12 x 3 blocks
  dim3 gD((TT + 127) / 128, DD / 64);        // 9 x 12 blocks
  dim3 gF((TT + 127) / 128, FFD / 64);       // 9 x 48 blocks

  for (int i = 0; i < NLAYERS; ++i) {
    QKVArgs qkv;
    qkv.B0 = Wq + (size_t)i * DD * DD;  qkv.bias0 = bq + i * DD;  qkv.C0 = qb;
    qkv.B1 = Wk + (size_t)i * DD * DD;  qkv.bias1 = bk + i * DD;  qkv.C1 = kb;
    qkv.B2 = Wv + (size_t)i * DD * DD;  qkv.bias2 = bv + i * DD;  qkv.C2 = vb;
    const float* w1 = W1 + (size_t)i * DD * FFD;
    const float* w2 = W2 + (size_t)i * FFD * DD;

    gemm_qkv<<<gQKV, blk, 0, stream>>>(xb, qkv, TT, DD, DD);

    attn_kernel<<<dim3(TT, HH), blk, 0, stream>>>(qb, kb, vb, cb);

    add_ln_kernel<<<dim3(TT), blk, 0, stream>>>(xb, cb, g1 + i * DD, be1 + i * DD, xb);

    gemm_bias_epi<<<gF, blk, 0, stream>>>(xb, w1, b1 + i * FFD, nullptr, hb, TT, DD, FFD, 1);
    gemm_bias_epi<<<gD, blk, 0, stream>>>(hb, w2, b2 + i * DD, xb, tb, TT, FFD, DD, 2);

    add_ln_kernel<<<dim3(TT), blk, 0, stream>>>(tb, nullptr, g2 + i * DD, be2 + i * DD, xb);
  }

  // Final gather x[key_pos[:, M:]] == contiguous rows [NMB, TT)
  hipMemcpyAsync(d_out, xb + (size_t)NMB * DD,
                 (size_t)NN * LL * DD * sizeof(float),
                 hipMemcpyDeviceToDevice, stream);
}
